// ReadoutLayerMultidim_67302137528982
// MI455X (gfx1250) — compile-verified, matrix operand order checked by reading, not run on hardware
//
#include <hip/hip_runtime.h>
#include <hip/hip_bf16.h>

// Problem constants (match reference)
#define B_   8
#define N_   128
#define F_   256
#define HID_ 256

typedef __attribute__((ext_vector_type(16))) __bf16 v16bf;
typedef __attribute__((ext_vector_type(8)))  float  v8f;

static __device__ __forceinline__ unsigned short f32_to_bf16_rne(float f) {
  unsigned int u = __float_as_uint(f);
  u += 0x7FFFu + ((u >> 16) & 1u);   // round-to-nearest-even
  return (unsigned short)(u >> 16);
}

// --- Pooled path (tiny): mean over nodes -> @Wp -> relu -> @W1a + b1 ---

__global__ void k_mean(const float* __restrict__ x, float* __restrict__ m) {
  int b = blockIdx.x, f = threadIdx.x;            // 8 blocks x 256 threads
  const float* p = x + (size_t)b * N_ * F_ + f;
  float s = 0.f;
  for (int n = 0; n < N_; ++n) s += p[n * F_];
  m[b * F_ + f] = s * (1.0f / N_);
}

__global__ void k_hp(const float* __restrict__ m, const float* __restrict__ Wp,
                     float* __restrict__ hpr) {
  int b = blockIdx.x, g = threadIdx.x;
  float s = 0.f;
  for (int f = 0; f < F_; ++f) s += m[b * F_ + f] * Wp[f * F_ + g];
  hpr[b * F_ + g] = fmaxf(s, 0.f);                // relu belongs to concat(feats)
}

__global__ void k_P(const float* __restrict__ hpr, const float* __restrict__ W1,
                    const float* __restrict__ b1, float* __restrict__ P) {
  int b = blockIdx.x, h = threadIdx.x;
  float s = b1[h];                                // fold b1 into P
  for (int g = 0; g < F_; ++g) s += hpr[b * F_ + g] * W1[g * HID_ + h];
  P[b * HID_ + h] = s;
}

// --- Convert relu(x) to bf16; pre-transpose W1 slices to [HID][F] bf16 ---

__global__ void k_relu_bf16(const float* __restrict__ x, unsigned short* __restrict__ Xh) {
  int i = blockIdx.x * blockDim.x + threadIdx.x;  // exactly B*N*F threads
  Xh[i] = f32_to_bf16_rne(fmaxf(x[i], 0.f));
}

__global__ void k_w1t(const float* __restrict__ W1,
                      unsigned short* __restrict__ WbT, unsigned short* __restrict__ WcT) {
  int h = blockIdx.x, k = threadIdx.x;            // 256 x 256
  WbT[h * F_ + k] = f32_to_bf16_rne(W1[(F_     + k) * HID_ + h]);
  WcT[h * F_ + k] = f32_to_bf16_rne(W1[(2 * F_ + k) * HID_ + h]);
}

// --- WMMA GEMMs: A = relu(x)@W1b, C = relu(x)@W1c ; M=1024, N=256, K=256 ---
// One wave per 16x16 tile. A fragment: lane lr in each half holds row M=lr,
// K-halves split across lane halves (lanes 0-15: K 0-15, lanes 16-31: K 16-31).
// B fragment mirrors A using the pre-transposed weights (column n contiguous).
__global__ __launch_bounds__(32) void k_gemm_wmma(
    const unsigned short* __restrict__ Xh,
    const unsigned short* __restrict__ WbT, const unsigned short* __restrict__ WcT,
    float* __restrict__ Arow, float* __restrict__ Crow) {
  const int lane = threadIdx.x;
  const int hw = lane >> 4;         // which K-half this lane carries
  const int lr = lane & 15;
  const int tm = blockIdx.x;        // 64 M-tiles
  const int tn = blockIdx.y;        // 16 N-tiles
  const unsigned short* WT  = (blockIdx.z == 0) ? WbT : WcT;
  float*                Out = (blockIdx.z == 0) ? Arow : Crow;

  const unsigned short* arow = Xh + (size_t)(tm * 16 + lr) * F_ + hw * 16;
  const unsigned short* brow = WT + (size_t)(tn * 16 + lr) * F_ + hw * 16;

  v8f acc = {};
#pragma unroll
  for (int k = 0; k < F_; k += 32) {
    v16bf a = *(const v16bf*)(arow + k);
    v16bf b = *(const v16bf*)(brow + k);
    acc = __builtin_amdgcn_wmma_f32_16x16x32_bf16(
        /*neg_a=*/false, a, /*neg_b=*/false, b,
        /*c_mod=*/(short)0, acc, /*reuse_a=*/false, /*reuse_b=*/false);
  }

  // D layout: VGPR r -> (M = 8*hw + r, Ncol = lr)
  float* orow = Out + (size_t)(tm * 16 + hw * 8) * HID_ + tn * 16 + lr;
#pragma unroll
  for (int r = 0; r < 8; ++r) orow[r * HID_] = acc[r];
}

// --- Pairwise epilogue: out[b,i,j] = relu(P[b]+A[b,i]+C[b,j]) . W2 + b2 ---
__global__ __launch_bounds__(128) void k_pair(
    const float* __restrict__ P, const float* __restrict__ Arow,
    const float* __restrict__ Crow, const float* __restrict__ W2,
    const float* __restrict__ b2, float* __restrict__ out) {
  __shared__ float baseS[HID_];
  __shared__ float w2S[HID_];
  __shared__ float cT[N_ * 33];     // stride-33 padding: kills 64-bank conflicts

  const int i = blockIdx.x, b = blockIdx.y, t = threadIdx.x;  // t == j

  for (int h = t; h < HID_; h += 128) {
    baseS[h] = P[b * HID_ + h] + Arow[(size_t)(b * N_ + i) * HID_ + h];
    w2S[h]   = W2[h];
  }

  float acc = b2[0];
  for (int h0 = 0; h0 < HID_; h0 += 32) {
    __syncthreads();                               // also covers baseS fill
    for (int e = t; e < N_ * 32; e += 128) {       // coalesced 128B rows
      int j = e >> 5, hh = e & 31;
      cT[j * 33 + hh] = Crow[(size_t)(b * N_ + j) * HID_ + h0 + hh];
    }
    __syncthreads();
#pragma unroll
    for (int hh = 0; hh < 32; ++hh) {
      float v = baseS[h0 + hh] + cT[t * 33 + hh];  // baseS: LDS broadcast
      acc = fmaf(fmaxf(v, 0.f), w2S[h0 + hh], acc);
    }
  }
  out[(size_t)(b * N_ + i) * N_ + t] = acc;
}

extern "C" void kernel_launch(void* const* d_in, const int* in_sizes, int n_in,
                              void* d_out, int out_size, void* d_ws, size_t ws_size,
                              hipStream_t stream) {
  (void)in_sizes; (void)n_in; (void)out_size; (void)ws_size;
  const float* x  = (const float*)d_in[0];
  // d_in[1] = mask (all ones; reference output does not use it)
  const float* Wp = (const float*)d_in[2];
  const float* W1 = (const float*)d_in[3];
  const float* b1 = (const float*)d_in[4];
  const float* W2 = (const float*)d_in[5];
  const float* b2 = (const float*)d_in[6];
  float* out = (float*)d_out;

  // Workspace layout (bytes), total ~2.78 MB
  char* ws = (char*)d_ws;
  float*          m    = (float*)(ws + 0);            //  8 KB  [B,F]
  float*          hpr  = (float*)(ws + 8192);         //  8 KB  [B,F]
  float*          P    = (float*)(ws + 16384);        //  8 KB  [B,HID]
  unsigned short* Xh   = (unsigned short*)(ws + 24576);   // 512 KB bf16 relu(x)
  unsigned short* WbT  = (unsigned short*)(ws + 548864);  // 128 KB bf16 W1b^T
  unsigned short* WcT  = (unsigned short*)(ws + 679936);  // 128 KB bf16 W1c^T
  float*          Arow = (float*)(ws + 811008);       // 1 MB  [B*N,HID]
  float*          Crow = (float*)(ws + 1859584);      // 1 MB  [B*N,HID]

  k_mean<<<B_, F_, 0, stream>>>(x, m);
  k_hp<<<B_, F_, 0, stream>>>(m, Wp, hpr);
  k_P<<<B_, HID_, 0, stream>>>(hpr, W1, b1, P);
  k_relu_bf16<<<(B_ * N_ * F_) / 256, 256, 0, stream>>>(x, Xh);
  k_w1t<<<HID_, F_, 0, stream>>>(W1, WbT, WcT);

  dim3 gg(64, 16, 2);               // M-tiles x N-tiles x {A,C}
  k_gemm_wmma<<<gg, 32, 0, stream>>>(Xh, WbT, WcT, Arow, Crow);

  dim3 gp(N_, B_);
  k_pair<<<gp, N_, 0, stream>>>(P, Arow, Crow, W2, b2, out);
}
